// SimpleTransformerBlock_3685081940658
// MI455X (gfx1250) — matrix operand order repaired and checked by reading, not compiled, not run on hardware
//
#include <hip/hip_runtime.h>

typedef __bf16 bf16_t;
typedef __attribute__((ext_vector_type(16))) __bf16 v16bf;
typedef __attribute__((ext_vector_type(8)))  __bf16 v8bf;
typedef __attribute__((ext_vector_type(8)))  float  v8f;

#define Bsz 2
#define Lseq 2048
#define Dh 1024
#define Hn 16
#define HD 64
#define Mlp 4096
#define MTOK 4096

__device__ __forceinline__ float bf2f(bf16_t h) { return (float)h; }
__device__ __forceinline__ bf16_t f2bf(float f) { return (bf16_t)f; }

// ---------------------------------------------------------------------------
// gfx1250 async global->LDS staging (ASYNCcnt-tracked DMA, bypasses VGPRs).
// LDS byte offset = low 32 bits of the generic pointer (ISA 10.2 aperture map).
// ---------------------------------------------------------------------------
__device__ __forceinline__ void async_ld_b128(void* lds, const void* gaddr) {
  unsigned loff = (unsigned)(size_t)lds;
  asm volatile("global_load_async_to_lds_b128 %0, %1, off"
               :: "v"(loff), "v"(gaddr) : "memory");
}
__device__ __forceinline__ void wait_async0() {
  asm volatile("s_wait_asynccnt 0" ::: "memory");
}

// ---------------------------------------------------------------------------
// 16-lane butterfly reductions. Prefer v_permlane16_b32 (VALU, no LDS
// traffic) over ds_bpermute-based __shfl_xor. Selectors: lane -> lane^k.
// ---------------------------------------------------------------------------
#if __has_builtin(__builtin_amdgcn_permlane16)
__device__ __forceinline__ float xor16(float v, unsigned lo, unsigned hi) {
  unsigned u = __builtin_bit_cast(unsigned, v);
  u = __builtin_amdgcn_permlane16(u, u, lo, hi, false, false);
  return __builtin_bit_cast(float, u);
}
__device__ __forceinline__ float rmax16(float v) {
  v = fmaxf(v, xor16(v, 0x67452301u, 0xEFCDAB89u));  // k=1
  v = fmaxf(v, xor16(v, 0x54761032u, 0xDCFE98BAu));  // k=2
  v = fmaxf(v, xor16(v, 0x32107654u, 0xBA98FEDCu));  // k=4
  v = fmaxf(v, xor16(v, 0xFEDCBA98u, 0x76543210u));  // k=8
  return v;
}
__device__ __forceinline__ float rsum16(float v) {
  v += xor16(v, 0x67452301u, 0xEFCDAB89u);
  v += xor16(v, 0x54761032u, 0xDCFE98BAu);
  v += xor16(v, 0x32107654u, 0xBA98FEDCu);
  v += xor16(v, 0xFEDCBA98u, 0x76543210u);
  return v;
}
#else
__device__ __forceinline__ float rmax16(float v) {
  for (int d = 1; d < 16; d <<= 1) v = fmaxf(v, __shfl_xor(v, d, 32));
  return v;
}
__device__ __forceinline__ float rsum16(float v) {
  for (int d = 1; d < 16; d <<= 1) v += __shfl_xor(v, d, 32);
  return v;
}
#endif

// ---------------------------------------------------------------------------
// Weight pack: fp32 row-major W[K][N]  ->  bf16 transposed WT[N][K]
// ---------------------------------------------------------------------------
__global__ __launch_bounds__(256) void pack_wT_kernel(
    const float* __restrict__ w, bf16_t* __restrict__ wT, int K, int N) {
  size_t i = (size_t)blockIdx.x * blockDim.x + threadIdx.x;
  size_t total = (size_t)K * N;
  if (i >= total) return;
  int n = (int)(i % N);
  int k = (int)(i / N);
  wT[(size_t)n * K + k] = f2bf(w[i]);
}

// ---------------------------------------------------------------------------
// LayerNorm: fp32 [token][D] -> bf16 [token][D].  One 256-thread block/token.
// ---------------------------------------------------------------------------
__global__ __launch_bounds__(256) void layernorm_kernel(
    const float* __restrict__ x, const float* __restrict__ scale,
    const float* __restrict__ bias, bf16_t* __restrict__ out) {
  int t = blockIdx.x;
  const float* xr = x + (size_t)t * Dh;
  float v[4];
  float sum = 0.f;
#pragma unroll
  for (int i = 0; i < 4; ++i) { v[i] = xr[threadIdx.x + i * 256]; sum += v[i]; }
  __shared__ float red[8];
  for (int d = 1; d < 32; d <<= 1) sum += __shfl_xor(sum, d, 32);
  int wave = threadIdx.x >> 5, lane = threadIdx.x & 31;
  if (lane == 0) red[wave] = sum;
  __syncthreads();
  float mu = 0.f;
#pragma unroll
  for (int w = 0; w < 8; ++w) mu += red[w];
  mu *= (1.f / Dh);
  float vs = 0.f;
#pragma unroll
  for (int i = 0; i < 4; ++i) { float d = v[i] - mu; vs += d * d; }
  for (int d = 1; d < 32; d <<= 1) vs += __shfl_xor(vs, d, 32);
  __syncthreads();
  if (lane == 0) red[wave] = vs;
  __syncthreads();
  float var = 0.f;
#pragma unroll
  for (int w = 0; w < 8; ++w) var += red[w];
  var *= (1.f / Dh);
  float rstd = rsqrtf(var + 1e-6f);
  bf16_t* orow = out + (size_t)t * Dh;
#pragma unroll
  for (int i = 0; i < 4; ++i) {
    int c = threadIdx.x + i * 256;
    orow[c] = f2bf((v[i] - mu) * rstd * scale[c] + bias[c]);
  }
}

// ---------------------------------------------------------------------------
// bf16 WMMA GEMM:  out[M][N] = A[M][K] @ (BT[N][K])^T + bias
// Block 128x128, 8 waves (4x2), 32x64 per wave (2x4 WMMA tiles).
// Double-buffered LDS, async global->LDS staging overlapped with compute.
//   MODE 0: QKV scatter  (q scaled 1/8, q/k: [b][h][l][hd], v: [b][h][hd][l])
//   MODE 1: f32 out + residual add
//   MODE 2: bf16 out + tanh-GELU
// ---------------------------------------------------------------------------
#define TM 128
#define TN 128
#define TK 32
#define TKP 48  // padded LDS row (96B): 16B-aligned, dodges bank conflicts

template <int MODE>
__global__ __launch_bounds__(256) void gemm_bf16_kernel(
    const bf16_t* __restrict__ A, const bf16_t* __restrict__ BT,
    const float* __restrict__ bias, const float* __restrict__ resid,
    float* __restrict__ outf, bf16_t* __restrict__ outb,
    bf16_t* __restrict__ qo, bf16_t* __restrict__ ko, bf16_t* __restrict__ vo,
    int M, int N, int K) {
  __shared__ bf16_t As[2][TM][TKP];
  __shared__ bf16_t Bs[2][TN][TKP];

  const int tid = threadIdx.x;
  const int wave = tid >> 5;
  const int lane = tid & 31;
  const int wm = wave >> 1;  // 0..3  (m tiles of 32)
  const int wn = wave & 1;   // 0..1  (n tiles of 64)
  const int bM = blockIdx.x * TM;
  const int bN = blockIdx.y * TN;
  const int half = lane >> 4;
  const int l16 = lane & 15;

  const int sr = tid >> 1;          // 0..127 staging row
  const int sk = (tid & 1) * 16;    // 0 or 16

  v8f acc[2][4] = {};
  const int nk = K / TK;

  // prologue: stage tile 0 into buffer 0
  {
    const bf16_t* ga = A + (size_t)(bM + sr) * K + sk;
    async_ld_b128(&As[0][sr][sk], ga);
    async_ld_b128(&As[0][sr][sk + 8], ga + 8);
    const bf16_t* gb = BT + (size_t)(bN + sr) * K + sk;
    async_ld_b128(&Bs[0][sr][sk], gb);
    async_ld_b128(&Bs[0][sr][sk + 8], gb + 8);
  }
  wait_async0();
  __syncthreads();

  for (int kt = 0; kt < nk; ++kt) {
    const int buf = kt & 1;
    if (kt + 1 < nk) {  // async-stage next tile into the other buffer
      const int k0 = (kt + 1) * TK;
      const bf16_t* ga = A + (size_t)(bM + sr) * K + k0 + sk;
      async_ld_b128(&As[buf ^ 1][sr][sk], ga);
      async_ld_b128(&As[buf ^ 1][sr][sk + 8], ga + 8);
      const bf16_t* gb = BT + (size_t)(bN + sr) * K + k0 + sk;
      async_ld_b128(&Bs[buf ^ 1][sr][sk], gb);
      async_ld_b128(&Bs[buf ^ 1][sr][sk + 8], gb + 8);
    }

    v16bf af[2], bfr[4];
#pragma unroll
    for (int mt = 0; mt < 2; ++mt) {
      const bf16_t* row = &As[buf][wm * 32 + mt * 16 + l16][0];
      v8bf lo = *(const v8bf*)(row + half * 8);
      v8bf hi = *(const v8bf*)(row + 16 + half * 8);
#pragma unroll
      for (int i = 0; i < 8; ++i) { af[mt][i] = lo[i]; af[mt][i + 8] = hi[i]; }
    }
#pragma unroll
    for (int nt = 0; nt < 4; ++nt) {
      const bf16_t* row = &Bs[buf][wn * 64 + nt * 16 + l16][0];
      v8bf lo = *(const v8bf*)(row + half * 8);
      v8bf hi = *(const v8bf*)(row + 16 + half * 8);
#pragma unroll
      for (int i = 0; i < 8; ++i) { bfr[nt][i] = lo[i]; bfr[nt][i + 8] = hi[i]; }
    }
#pragma unroll
    for (int mt = 0; mt < 2; ++mt)
#pragma unroll
      for (int nt = 0; nt < 4; ++nt)
        acc[mt][nt] = __builtin_amdgcn_wmma_f32_16x16x32_bf16(
            false, af[mt], false, bfr[nt], (short)0, acc[mt][nt], false, false);

    wait_async0();     // next tile has landed in LDS
    __syncthreads();   // all waves done reading current buffer
  }

  // Epilogue. D layout: lane -> n = l16 (+tile), m = r + 8*half (+tile).
#pragma unroll
  for (int mt = 0; mt < 2; ++mt) {
#pragma unroll
    for (int nt = 0; nt < 4; ++nt) {
      int col = bN + wn * 64 + nt * 16 + l16;
      float bval = bias[col];
#pragma unroll
      for (int r = 0; r < 8; ++r) {
        int row = bM + wm * 32 + mt * 16 + r + half * 8;
        float v = acc[mt][nt][r] + bval;
        if (MODE == 0) {
          int which = col >> 10;  // 0=q 1=k 2=v
          int c = col & 1023;
          int hh = c >> 6, hd = c & 63;
          int b = row >> 11, l = row & 2047;
          size_t bh = (size_t)(b * Hn + hh);
          if (which == 0) {
            qo[(bh * Lseq + l) * HD + hd] = f2bf(v * 0.125f);  // 1/sqrt(64)
          } else if (which == 1) {
            ko[(bh * Lseq + l) * HD + hd] = f2bf(v);
          } else {
            vo[(bh * HD + hd) * Lseq + l] = f2bf(v);  // transposed
          }
        } else if (MODE == 1) {
          size_t idx = (size_t)row * N + col;
          outf[idx] = v + resid[idx];
        } else {  // MODE 2: GELU (tanh approx) -> bf16
          float g = 0.5f * v *
                    (1.f + tanhf(0.7978845608028654f * (v + 0.044715f * v * v * v)));
          outb[(size_t)row * N + col] = f2bf(g);
        }
      }
    }
  }
}

// ---------------------------------------------------------------------------
// Flash attention: one wave per (b, h, 16-query tile). 128 threads = 4 waves.
// scores: A=q[16xHD], B=k rows (contiguous hd).  PV: A=P (via LDS relayout),
// B=vT rows (contiguous kv).  Online softmax across the n (=kv) lanes.
// ---------------------------------------------------------------------------
__global__ __launch_bounds__(128) void attn_kernel(
    const bf16_t* __restrict__ q, const bf16_t* __restrict__ k,
    const bf16_t* __restrict__ vT, bf16_t* __restrict__ o) {
  __shared__ bf16_t P[4][16][32];
  const int wave = threadIdx.x >> 5;
  const int lane = threadIdx.x & 31;
  const int gw = blockIdx.x * 4 + wave;  // 0 .. B*H*(L/16)-1
  const int qt = gw & 127;               // L/16 = 128
  const int bh = gw >> 7;                // 0..31
  const int half = lane >> 4, l16 = lane & 15;

  const bf16_t* qp = q + ((size_t)bh * Lseq + qt * 16) * HD;
  const bf16_t* kp = k + (size_t)bh * Lseq * HD;
  const bf16_t* vp = vT + (size_t)bh * HD * Lseq;

  v16bf qf[2];
#pragma unroll
  for (int kk = 0; kk < 2; ++kk) {
    const bf16_t* row = qp + (size_t)l16 * HD + kk * 32;
    v8bf lo = *(const v8bf*)(row + half * 8);
    v8bf hi = *(const v8bf*)(row + 16 + half * 8);
#pragma unroll
    for (int i = 0; i < 8; ++i) { qf[kk][i] = lo[i]; qf[kk][i + 8] = hi[i]; }
  }

  float mrow[8], lrow[8];
#pragma unroll
  for (int r = 0; r < 8; ++r) { mrow[r] = -1e30f; lrow[r] = 0.f; }
  v8f acc[4] = {};

  for (int j = 0; j < Lseq; j += 32) {
    v8f s[2] = {};
#pragma unroll
    for (int st = 0; st < 2; ++st) {
#pragma unroll
      for (int kk = 0; kk < 2; ++kk) {
        const bf16_t* row =
            kp + (size_t)(j + st * 16 + l16) * HD + kk * 32 + half * 8;
        v8bf lo = *(const v8bf*)row;
        v8bf hi = *(const v8bf*)(row + 16);
        v16bf kf;
#pragma unroll
        for (int i = 0; i < 8; ++i) { kf[i] = lo[i]; kf[i + 8] = hi[i]; }
        s[st] = __builtin_amdgcn_wmma_f32_16x16x32_bf16(
            false, qf[kk], false, kf, (short)0, s[st], false, false);
      }
    }
    float alpha[8];
#pragma unroll
    for (int r = 0; r < 8; ++r) {
      float mx = rmax16(fmaxf(s[0][r], s[1][r]));
      float mnew = fmaxf(mrow[r], mx);
      alpha[r] = __expf(mrow[r] - mnew);
      float p0 = __expf(s[0][r] - mnew);
      float p1 = __expf(s[1][r] - mnew);
      s[0][r] = p0; s[1][r] = p1;
      lrow[r] = lrow[r] * alpha[r] + rsum16(p0 + p1);
      mrow[r] = mnew;
    }
#pragma unroll
    for (int r = 0; r < 8; ++r) {
      int m = r + 8 * half;
      P[wave][m][l16] = f2bf(s[0][r]);
      P[wave][m][16 + l16] = f2bf(s[1][r]);
    }
#pragma unroll
    for (int nt = 0; nt < 4; ++nt)
#pragma unroll
      for (int r = 0; r < 8; ++r) acc[nt][r] *= alpha[r];

    v16bf pf;
    {
      const bf16_t* row = &P[wave][l16][0];
      v8bf lo = *(const v8bf*)(row + half * 8);
      v8bf hi = *(const v8bf*)(row + 16 + half * 8);
#pragma unroll
      for (int i = 0; i < 8; ++i) { pf[i] = lo[i]; pf[i + 8] = hi[i]; }
    }
#pragma unroll
    for (int nt = 0; nt < 4; ++nt) {
      const bf16_t* row = vp + (size_t)(nt * 16 + l16) * Lseq + j + half * 8;
      v8bf lo = *(const v8bf*)row;
      v8bf hi = *(const v8bf*)(row + 16);
      v16bf vf;
#pragma unroll
      for (int i = 0; i < 8; ++i) { vf[i] = lo[i]; vf[i + 8] = hi[i]; }
      acc[nt] = __builtin_amdgcn_wmma_f32_16x16x32_bf16(
          false, pf, false, vf, (short)0, acc[nt], false, false);
    }
  }

  const int b = bh >> 4, hh = bh & 15;
#pragma unroll
  for (int nt = 0; nt < 4; ++nt) {
#pragma unroll
    for (int r = 0; r < 8; ++r) {
      int m = r + 8 * half;
      size_t idx = ((size_t)(b * Lseq + qt * 16 + m)) * Dh + hh * HD + nt * 16 + l16;
      o[idx] = f2bf(acc[nt][r] / lrow[r]);
    }
  }
}

// ---------------------------------------------------------------------------
extern "C" void kernel_launch(void* const* d_in, const int* in_sizes, int n_in,
                              void* d_out, int out_size, void* d_ws,
                              size_t ws_size, hipStream_t stream) {
  const float* x     = (const float*)d_in[0];
  const float* ln1_s = (const float*)d_in[1];
  const float* ln1_b = (const float*)d_in[2];
  const float* w_qkv = (const float*)d_in[3];
  const float* b_qkv = (const float*)d_in[4];
  const float* w_o   = (const float*)d_in[5];
  const float* b_o   = (const float*)d_in[6];
  const float* ln2_s = (const float*)d_in[7];
  const float* ln2_b = (const float*)d_in[8];
  const float* w_m1  = (const float*)d_in[9];
  const float* b_m1  = (const float*)d_in[10];
  const float* w_m2  = (const float*)d_in[11];
  const float* b_m2  = (const float*)d_in[12];

  char* ws = (char*)d_ws;
  size_t off = 0;
  auto alloc = [&](size_t bytes) -> void* {
    void* p = ws + off;
    off = (off + bytes + 255) & ~(size_t)255;
    return p;
  };
  bf16_t* wT_qkv = (bf16_t*)alloc((size_t)3 * Dh * Dh * 2);
  bf16_t* wT_o   = (bf16_t*)alloc((size_t)Dh * Dh * 2);
  bf16_t* wT_m1  = (bf16_t*)alloc((size_t)Mlp * Dh * 2);
  bf16_t* wT_m2  = (bf16_t*)alloc((size_t)Dh * Mlp * 2);
  bf16_t* h1     = (bf16_t*)alloc((size_t)MTOK * Dh * 2);
  bf16_t* qb     = (bf16_t*)alloc((size_t)MTOK * Dh * 2);
  bf16_t* kb     = (bf16_t*)alloc((size_t)MTOK * Dh * 2);
  bf16_t* vTb    = (bf16_t*)alloc((size_t)MTOK * Dh * 2);
  bf16_t* ob     = (bf16_t*)alloc((size_t)MTOK * Dh * 2);
  float*  x2     = (float*)alloc((size_t)MTOK * Dh * 4);
  bf16_t* h2     = (bf16_t*)alloc((size_t)MTOK * Dh * 2);
  bf16_t* mlp    = (bf16_t*)alloc((size_t)MTOK * Mlp * 2);

  // 1) pack weights (fp32 row-major -> bf16 transposed)
  {
    size_t n;
    n = (size_t)Dh * 3 * Dh;
    pack_wT_kernel<<<(n + 255) / 256, 256, 0, stream>>>(w_qkv, wT_qkv, Dh, 3 * Dh);
    n = (size_t)Dh * Dh;
    pack_wT_kernel<<<(n + 255) / 256, 256, 0, stream>>>(w_o, wT_o, Dh, Dh);
    n = (size_t)Dh * Mlp;
    pack_wT_kernel<<<(n + 255) / 256, 256, 0, stream>>>(w_m1, wT_m1, Dh, Mlp);
    n = (size_t)Mlp * Dh;
    pack_wT_kernel<<<(n + 255) / 256, 256, 0, stream>>>(w_m2, wT_m2, Mlp, Dh);
  }

  // 2) LN1
  layernorm_kernel<<<MTOK, 256, 0, stream>>>(x, ln1_s, ln1_b, h1);

  // 3) QKV projection (scatter epilogue)
  {
    dim3 grid(MTOK / TM, (3 * Dh) / TN);
    gemm_bf16_kernel<0><<<grid, 256, 0, stream>>>(
        h1, wT_qkv, b_qkv, nullptr, nullptr, nullptr, qb, kb, vTb,
        MTOK, 3 * Dh, Dh);
  }

  // 4) flash attention
  attn_kernel<<<(Bsz * Hn * (Lseq / 16)) / 4, 128, 0, stream>>>(qb, kb, vTb, ob);

  // 5) output projection + residual -> x2 (f32)
  {
    dim3 grid(MTOK / TM, Dh / TN);
    gemm_bf16_kernel<1><<<grid, 256, 0, stream>>>(
        ob, wT_o, b_o, x, x2, nullptr, nullptr, nullptr, nullptr,
        MTOK, Dh, Dh);
  }

  // 6) LN2
  layernorm_kernel<<<MTOK, 256, 0, stream>>>(x2, ln2_s, ln2_b, h2);

  // 7) MLP1 + GELU -> bf16
  {
    dim3 grid(MTOK / TM, Mlp / TN);
    gemm_bf16_kernel<2><<<grid, 256, 0, stream>>>(
        h2, wT_m1, b_m1, nullptr, nullptr, mlp, nullptr, nullptr, nullptr,
        MTOK, Mlp, Dh);
  }

  // 8) MLP2 + residual -> d_out (f32)
  {
    dim3 grid(MTOK / TM, Dh / TN);
    gemm_bf16_kernel<1><<<grid, 256, 0, stream>>>(
        mlp, wT_m2, b_m2, x2, (float*)d_out, nullptr, nullptr, nullptr, nullptr,
        MTOK, Dh, Mlp);
  }
}
